// ContinuousHopfieldNet_70652212019686
// MI455X (gfx1250) — matrix operand order repaired and checked by reading, not compiled, not run on hardware
//
#include <hip/hip_runtime.h>
#include <hip/hip_bf16.h>

typedef __attribute__((ext_vector_type(16))) _Float16 v16h;
typedef __attribute__((ext_vector_type(8)))  _Float16 v8h;
typedef __attribute__((ext_vector_type(8)))  float    v8f;
typedef int v4i_gcc __attribute__((vector_size(16)));   // matches builtin pointee

#define NB      1024   // basis functions / bins
#define DQ      1024   // feature dim D
#define NQ_     1024   // number of queries
#define NPTS    2048   // quadrature points
#define BETA_   1.0f

// GEMM tiling
#define TM      64     // block M tile
#define TN      128    // block N tile
#define KSTEP   32     // K per v_wmma_f32_16x16x32_f16
#define LPITCH  40     // halfs per LDS row (80B: 16B-aligned, 20-bank stride)

// ---------------------------------------------------------------------------
// CDNA5 async global->LDS path (ASYNCcnt), with sync fallback.
// Builtin signature (probe-verified): (int4 AS1*, int4 AS3*, Ii offset, Ii cpol)
// ---------------------------------------------------------------------------
#if __has_builtin(__builtin_amdgcn_global_load_async_to_lds_b128)
#define HAVE_ASYNC_LDS 1
#else
#define HAVE_ASYNC_LDS 0
#endif

__device__ __forceinline__ void copy16_to_lds(const _Float16* src, _Float16* dst) {
#if HAVE_ASYNC_LDS
    __builtin_amdgcn_global_load_async_to_lds_b128(
        (__attribute__((address_space(1))) v4i_gcc*)(void*)src,
        (__attribute__((address_space(3))) v4i_gcc*)(void*)dst,
        0, 0);
#else
    *(v8h*)dst = *(const v8h*)src;
#endif
}

__device__ __forceinline__ void wait_async_all() {
#if HAVE_ASYNC_LDS
#if __has_builtin(__builtin_amdgcn_s_wait_asynccnt)
    __builtin_amdgcn_s_wait_asynccnt(0);
#else
    asm volatile("s_wait_asynccnt 0x0" ::: "memory");
#endif
#endif
}

// ---------------------------------------------------------------------------
// B[b,d] = (k[4b,d]+k[4b+1,d]+k[4b+2,d]+k[4b+3,d]) / 4.5   (A = FF^T+0.5I = 4.5I)
// Emit f16 row-major Bh and f16 transpose BhT (so both GEMMs are "NT" form).
// ---------------------------------------------------------------------------
__global__ void build_B_kernel(const float* __restrict__ k,
                               _Float16* __restrict__ Bh,
                               _Float16* __restrict__ BhT) {
    int idx = blockIdx.x * blockDim.x + threadIdx.x;   // 0 .. NB*DQ-1
    int b = idx >> 10;
    int d = idx & (DQ - 1);
    const float inv = 1.0f / 4.5f;
    size_t base = (size_t)(4 * b) * DQ + d;
    float s = k[base] + k[base + DQ] + k[base + 2 * DQ] + k[base + 3 * DQ];
    s *= inv;
    _Float16 h = (_Float16)s;
    Bh [(size_t)b * DQ + d] = h;
    BhT[(size_t)d * NB + b] = h;
}

__global__ void convert_q_kernel(const float* __restrict__ q,
                                 _Float16* __restrict__ Qh) {
    int idx = blockIdx.x * blockDim.x + threadIdx.x;
    Qh[idx] = (_Float16)q[idx];
}

// ---------------------------------------------------------------------------
// C(MxN) = A(MxK) * Bt(NxK)^T, M=N=K=1024, f16 operands, f32 accum.
// Block = 256 threads (8 waves) computes a 64x128 C tile; waves arranged 4x2,
// each wave owns 16x64 (one A fragment reused across 4 WMMA per K-step).
// K-tiles staged to LDS (double-buffered) via async global->LDS loads.
// Fragment layouts per CDNA5 ISA 7.12.2 (wave32):
//   A: lane l (m=l&15): e0..7 -> K=kb..kb+7, e8..15 -> K=kb+16..kb+23, kb=(l>=16)*8
//   B: lane l (n=l&15): e0..7 -> K=kb..kb+7, e8..15 -> K=kb+16..kb+23, kb=(l>=16)*8
//   C: lane l (n=l&15): elem r -> M = r + (l>=16)*8
// ---------------------------------------------------------------------------
__global__ void __launch_bounds__(256)
wmma_gemm_nt_kernel(const _Float16* __restrict__ A,
                    const _Float16* __restrict__ Bt,
                    float* __restrict__ Cf,        // may be null
                    _Float16* __restrict__ Ch) {   // may be null
    __shared__ _Float16 sA[2][TM * LPITCH];        // 2 x 2560 halfs
    __shared__ _Float16 sB[2][TN * LPITCH];        // 2 x 5120 halfs

    const int tid  = threadIdx.x;
    const int lane = tid & 31;
    const int wave = tid >> 5;
    const int bm = blockIdx.x & 15;                // 16 m-blocks of 64
    const int bn = blockIdx.x >> 4;                // 8 n-blocks of 128
    const int mbase = bm * TM;
    const int nbase = bn * TN;
    const int wm = (wave & 3) * 16;                // wave m-subtile in block
    const int wn = (wave >> 2) * 64;               // wave n-range start

    const int mr  = lane & 15;
    const int hi  = lane >> 4;                     // 0/1
    const int kbA = hi << 3;                       // 0 or 8 halfs
    const int kbB = hi << 3;                       // 0 or 8 halfs

    v8f acc[4] = {v8f{}, v8f{}, v8f{}, v8f{}};

    // Branch-free staging: thread's 3 chunks map to A[0..63], Bt[0..63],
    // Bt[64..127] rows respectively (16B per chunk).
    const int srow = tid >> 2;                     // 0..63
    const int sqw  = (tid & 3) * 8;                // half offset in row
    auto stage = [&](int k0, int buf) {
        copy16_to_lds(A  + (size_t)(mbase + srow) * DQ + k0 + sqw,
                      &sA[buf][srow * LPITCH + sqw]);
        copy16_to_lds(Bt + (size_t)(nbase + srow) * DQ + k0 + sqw,
                      &sB[buf][srow * LPITCH + sqw]);
        copy16_to_lds(Bt + (size_t)(nbase + 64 + srow) * DQ + k0 + sqw,
                      &sB[buf][(64 + srow) * LPITCH + sqw]);
    };

    stage(0, 0);
    wait_async_all();
    __syncthreads();

    union Frag { v16h v; v8h h[2]; };

    for (int kt = 0; kt < DQ / KSTEP; ++kt) {
        const int cur = kt & 1;
        if (kt + 1 < DQ / KSTEP) stage((kt + 1) * KSTEP, cur ^ 1);

        // Preload A + all 4 B fragments, then issue 4 back-to-back WMMAs.
        const _Float16* arow = &sA[cur][(wm + mr) * LPITCH];
        Frag af;
        af.h[0] = *(const v8h*)(arow + kbA);          // K = kb..kb+7
        af.h[1] = *(const v8h*)(arow + kbA + 16);     // K = kb+16..kb+23

        Frag bf[4];
        #pragma unroll
        for (int j = 0; j < 4; ++j) {
            const _Float16* brow = &sB[cur][(wn + j * 16 + mr) * LPITCH];
            bf[j].h[0] = *(const v8h*)(brow + kbB);      // K = kb..kb+7
            bf[j].h[1] = *(const v8h*)(brow + kbB + 16); // K = kb+16..kb+23
        }

        #pragma unroll
        for (int j = 0; j < 4; ++j)
            acc[j] = __builtin_amdgcn_wmma_f32_16x16x32_f16(
                false, af.v, false, bf[j].v, (short)0, acc[j], false, false);

        wait_async_all();      // next tile landed in LDS
        __syncthreads();       // all waves done reading current buffer
    }

    const int mb2 = hi << 3;   // 0 or 8
    #pragma unroll
    for (int j = 0; j < 4; ++j) {
        const size_t col = (size_t)(nbase + wn + j * 16 + mr);
        if (Cf) {
            #pragma unroll
            for (int r = 0; r < 8; ++r)
                Cf[(size_t)(mbase + wm + mb2 + r) * 1024 + col] = acc[j][r];
        }
        if (Ch) {
            #pragma unroll
            for (int r = 0; r < 8; ++r)
                Ch[(size_t)(mbase + wm + mb2 + r) * 1024 + col] = (_Float16)acc[j][r];
        }
    }
}

// ---------------------------------------------------------------------------
// Per query row n:
//   score_i = BETA * S0[n, bin(t_i)]   (0 if t_i lands in no bin, e.g. t=1.0)
//   m = max_i score_i ; e_i = exp(score_i - m) ; Z = sum_i e_i * w_i
//   integral[n,b] = (1/Z) * sum_{i: bin(i)=b} e_i * w_i     -> Ph (f16)
// ---------------------------------------------------------------------------
__global__ void softmax_integral_kernel(const float* __restrict__ S0,
                                        _Float16* __restrict__ Ph) {
    __shared__ float sS[NB];
    __shared__ float sInt[NB];
    __shared__ float red[8];
    __shared__ float sBcast;

    const int row = blockIdx.x;
    const int tid = threadIdx.x;
    const int lane = tid & 31;
    const int wv = tid >> 5;
    const float dt = 1.0f / (float)(NPTS - 1);

    #pragma unroll
    for (int j = 0; j < NB / 256; ++j) {
        sS[tid + 256 * j]   = S0[(size_t)row * NB + tid + 256 * j];
        sInt[tid + 256 * j] = 0.0f;
    }
    __syncthreads();

    float sc[8];
    int   bn[8];
    float wt[8];
    #pragma unroll
    for (int j = 0; j < 8; ++j) {
        int i = tid + 256 * j;
        float t = (float)i * dt;
        int b = (int)(t * (float)NB);
        bool valid = (b < NB) && (t >= (float)b / (float)NB) &&
                     (t < (float)(b + 1) / (float)NB);
        bn[j] = valid ? b : -1;
        sc[j] = valid ? BETA_ * sS[b] : 0.0f;
        wt[j] = (i == 0 || i == NPTS - 1) ? 0.5f * dt : dt;
    }

    // ---- max reduction ----
    float m = sc[0];
    #pragma unroll
    for (int j = 1; j < 8; ++j) m = fmaxf(m, sc[j]);
    #pragma unroll
    for (int off = 16; off > 0; off >>= 1)
        m = fmaxf(m, __shfl_xor(m, off, 32));
    if (lane == 0) red[wv] = m;
    __syncthreads();
    if (tid == 0) {
        float mm = red[0];
        #pragma unroll
        for (int w = 1; w < 8; ++w) mm = fmaxf(mm, red[w]);
        sBcast = mm;
    }
    __syncthreads();
    m = sBcast;
    __syncthreads();

    // ---- exp, Z reduction, scatter e*w into bins (ds_add_f32) ----
    float zp = 0.0f;
    #pragma unroll
    for (int j = 0; j < 8; ++j) {
        float e  = __expf(sc[j] - m);
        float ew = e * wt[j];
        zp += ew;
        if (bn[j] >= 0) atomicAdd(&sInt[bn[j]], ew);
    }
    #pragma unroll
    for (int off = 16; off > 0; off >>= 1)
        zp += __shfl_xor(zp, off, 32);
    if (lane == 0) red[wv] = zp;
    __syncthreads();
    if (tid == 0) {
        float zz = 0.0f;
        #pragma unroll
        for (int w = 0; w < 8; ++w) zz += red[w];
        sBcast = zz;
    }
    __syncthreads();
    const float invZ = 1.0f / sBcast;

    #pragma unroll
    for (int j = 0; j < NB / 256; ++j) {
        int b = tid + 256 * j;
        Ph[(size_t)row * NB + b] = (_Float16)(sInt[b] * invZ);
    }
}

// ---------------------------------------------------------------------------
// ws layout (bytes):
//   [0,2M)   Qh   f16 1024x1024
//   [2M,4M)  Bh   f16 1024x1024
//   [4M,6M)  BhT  f16 1024x1024
//   [6M,8M)  Ph   f16 1024x1024
//   [8M,12M) S0   f32 1024x1024
// ---------------------------------------------------------------------------
extern "C" void kernel_launch(void* const* d_in, const int* in_sizes, int n_in,
                              void* d_out, int out_size, void* d_ws, size_t ws_size,
                              hipStream_t stream) {
    const float* k = (const float*)d_in[0];   // (4096, 1024) f32
    const float* q = (const float*)d_in[1];   // (1024, 1024) f32
    float* out = (float*)d_out;               // (1024, 1024) f32
    char* ws = (char*)d_ws;

    _Float16* Qh  = (_Float16*)(ws);
    _Float16* Bh  = (_Float16*)(ws + (2ull << 20));
    _Float16* BhT = (_Float16*)(ws + (4ull << 20));
    _Float16* Ph  = (_Float16*)(ws + (6ull << 20));
    float*    S0  = (float*)   (ws + (8ull << 20));

    build_B_kernel  <<<(NB * DQ) / 256, 256, 0, stream>>>(k, Bh, BhT);
    convert_q_kernel<<<(NQ_ * DQ) / 256, 256, 0, stream>>>(q, Qh);

    for (int it = 0; it < 3; ++it) {
        // S0 = Q @ B^T
        wmma_gemm_nt_kernel<<<128, 256, 0, stream>>>(Qh, Bh, S0, (_Float16*)nullptr);
        // softmax over 2048 points -> integral coefficients Ph
        softmax_integral_kernel<<<NQ_, 256, 0, stream>>>(S0, Ph);
        // Qnew = Ph @ B (via BhT rows); final iteration writes f32 output
        wmma_gemm_nt_kernel<<<128, 256, 0, stream>>>(
            Ph, BhT, (it == 2) ? out : (float*)nullptr, Qh);
    }
}
